// Qwen3NextGatedDeltaNetBase_82540681494770
// MI455X (gfx1250) — matrix operand order repaired and checked by reading, not compiled, not run on hardware
//
#include <hip/hip_runtime.h>
#include <hip/hip_bf16.h>

// ---------------------------------------------------------------------------
// Qwen3-Next Gated DeltaNet forward for MI455X (gfx1250, wave32, WMMA).
//
// Kernel 1 (memory-bound, L2-resident): depthwise causal conv1d(K=4) + SiLU,
//   split q/k/v, per-head L2 norm of q/k (q * DK^-0.5), gates. q/k are written
//   to the workspace as bf16 (consumed by 4 scan blocks each -> convert once,
//   half the bytes); v stays f32 (feeds f32 delta rhs).
//
// Kernel 2 (WMMA-bound): chunked delta-rule scan, chunk C=32, per-block state
//   S[128x64] f32 resident in LDS across all 64 chunks. All heavy math is
//   v_wmma_f32_16x16x32_bf16 (bf16 operands, f32 accumulate). All LDS buffers
//   feeding WMMA operands are laid out so each lane's fragment slice is a
//   contiguous, 16B-aligned run -> ds_load_b128 / ds_store_b128:
//     - A operands: row-major, K contiguous (two 16B runs per lane)
//     - B operands: stored transposed [n][K] (one 32B run per lane)
//     - C/D operands: stored transposed [col][row] (one 32B run per lane)
//   Next chunk's q/k/v is prefetched (global_prefetch_b8) a full chunk ahead.
// ---------------------------------------------------------------------------

#define B_    2
#define T_    2048
#define HK_   16
#define HV_   32
#define DK_   128
#define DV_   128
#define QKV_  8192   // 2*HK*DK + HV*DV

typedef __attribute__((ext_vector_type(16))) __bf16          v16bf;
typedef __attribute__((ext_vector_type(16))) unsigned short  v16u;
typedef __attribute__((ext_vector_type(8)))  unsigned short  v8u;
typedef __attribute__((ext_vector_type(8)))  float           v8f;

static __device__ inline unsigned short f2bf(float f) {
    unsigned u = __float_as_uint(f);
    u += 0x7FFFu + ((u >> 16) & 1u);          // round-to-nearest-even
    return (unsigned short)(u >> 16);
}
static __device__ inline float bf2f(unsigned short h) {
    return __uint_as_float(((unsigned)h) << 16);
}

static __device__ inline v8f wmma_bf16(v16bf a, v16bf b, v8f c) {
    // v_wmma_f32_16x16x32_bf16: D = A(16x32) * B(32x16) + C(16x16 f32)
    return __builtin_amdgcn_wmma_f32_16x16x32_bf16(
        false, a, false, b, (short)0, c, false, false);
}

// A fragment (16x32 bf16) from row-major buf[rows][ld], tile (m0,k0).
// Lane layout: m=lane&15, half=lane>>4; element e -> K=half*16+e.
// Two contiguous 8-element (16B) runs at +0 and +8 elements.
static __device__ inline v16bf load_a(const unsigned short* buf, int ld,
                                      int m0, int k0, int lane) {
    int m = m0 + (lane & 15);
    int half = lane >> 4;
    const unsigned short* p = buf + m * ld + k0 + half * 16;
    v8u lo = *(const v8u*)p;
    v8u hi = *(const v8u*)(p + 8);
    v16u f = __builtin_shufflevector(lo, hi, 0,1,2,3,4,5,6,7,8,9,10,11,12,13,14,15);
    return __builtin_bit_cast(v16bf, f);
}

// B fragment (32x16 bf16) from a TRANSPOSED buffer bufT[n][K] (ldT = K-extent).
// Lane layout: n=lane&15 (+n0), khalf=lane>>4, element e -> K = khalf*16+e:
// 16 contiguous elements (32B) per lane.
static __device__ inline v16bf load_b_tr(const unsigned short* bufT, int ldT,
                                         int k0, int n0, int lane) {
    int n = n0 + (lane & 15);
    int khalf = lane >> 4;
    const unsigned short* p = bufT + n * ldT + k0 + khalf * 16;
    return __builtin_bit_cast(v16bf, *(const v16u*)p);
}

// C/D f32 fragment from a TRANSPOSED buffer bufT[col][row] (ldT = row-extent).
static __device__ inline v8f load_cT(const float* bufT, int ldT,
                                     int m0, int n0, int lane) {
    int col = n0 + (lane & 15);
    int rbase = m0 + ((lane >> 4) ? 8 : 0);
    return *(const v8f*)(bufT + col * ldT + rbase);
}
static __device__ inline void store_cT(float* bufT, int ldT,
                                       int m0, int n0, int lane, v8f d) {
    int col = n0 + (lane & 15);
    int rbase = m0 + ((lane >> 4) ? 8 : 0);
    *(v8f*)(bufT + col * ldT + rbase) = d;
}

// Row-major C store (only for the small 32x32 Grams, consumed row-wise later).
static __device__ inline void store_c_rm(float* buf, int ld,
                                         int m0, int n0, int lane, v8f d) {
    int col = n0 + (lane & 15);
    int rbase = m0 + ((lane >> 4) ? 8 : 0);
#pragma unroll
    for (int r = 0; r < 8; ++r) buf[(rbase + r) * ld + col] = d[r];
}

// ---------------------------------------------------------------------------
// Kernel 1: conv + SiLU + split + l2norm + gates.  One block per (b,t).
// ---------------------------------------------------------------------------
__global__ __launch_bounds__(256)
void gdn_preprocess(const float* __restrict__ x,     // (B,T,QKV)
                    const float* __restrict__ bin,   // (B,T,HV)
                    const float* __restrict__ ain,   // (B,T,HV)
                    const float* __restrict__ cw,    // (QKV,4)
                    const float* __restrict__ dtb,   // (HV)
                    const float* __restrict__ alog,  // (HV)
                    unsigned short* __restrict__ qb, // (B,T,HK,DK) bf16, normed*DK^-0.5
                    unsigned short* __restrict__ kb, // (B,T,HK,DK) bf16, normed
                    float* __restrict__ vv,          // (B,T,HV,DV)
                    float* __restrict__ gg,          // (B,T,HV)
                    float* __restrict__ bb)          // (B,T,HV)
{
    __shared__ float yqk[2 * HK_ * DK_];   // 4096: staged q|k pre-norm
    __shared__ float red[32][8];
    __shared__ float scl[32];

    const int bt = blockIdx.x;            // b*T + t
    const int t  = bt % T_;
    const int b  = bt / T_;
    const float* xb = x + (size_t)b * T_ * QKV_;

    for (int c = threadIdx.x; c < QKV_; c += 256) {
        float acc = 0.f;
#pragma unroll
        for (int j = 0; j < 4; ++j) {
            int tt = t - 3 + j;
            if (tt >= 0) acc += xb[(size_t)tt * QKV_ + c] * cw[c * 4 + j];
        }
        float y = acc / (1.f + __expf(-acc));            // SiLU
        if (c < 4096) yqk[c] = y;
        else vv[(size_t)bt * (HV_ * DV_) + (c - 4096)] = y;
    }
    __syncthreads();

    {   // sum of squares for 32 sections (16 q-heads + 16 k-heads) of 128
        int s = threadIdx.x >> 3, l = threadIdx.x & 7;
        float p = 0.f;
        for (int j = l; j < 128; j += 8) { float u = yqk[s * 128 + j]; p += u * u; }
        red[s][l] = p;
    }
    __syncthreads();
    if (threadIdx.x < 32) {
        int s = threadIdx.x;
        float sm = 0.f;
#pragma unroll
        for (int l = 0; l < 8; ++l) sm += red[s][l];
        scl[s] = rsqrtf(sm + 1e-6f);
        // gates for head h = s
        float av = ain[(size_t)bt * HV_ + s] + dtb[s];
        float sp = (av > 20.f) ? av : log1pf(__expf(av));
        gg[(size_t)bt * HV_ + s] = -__expf(alog[s]) * sp;
        bb[(size_t)bt * HV_ + s] = 1.f / (1.f + __expf(-bin[(size_t)bt * HV_ + s]));
    }
    __syncthreads();

    const float qsc = 0.08838834764831845f;   // DK^-0.5
    for (int c = threadIdx.x; c < 4096; c += 256) {
        int s = c >> 7, d = c & 127;
        float val = yqk[c] * scl[s];
        if (s < 16) qb[(size_t)bt * 2048 + s * 128 + d] = f2bf(val * qsc);
        else        kb[(size_t)bt * 2048 + (s - 16) * 128 + d] = f2bf(val);
    }
}

// ---------------------------------------------------------------------------
// Kernel 2: chunked delta-rule scan. Grid = B*HV*2 = 128 blocks (dv split in 2),
// 256 threads = 8 wave32s. ~103 KB dynamic LDS, state resident across chunks.
// ---------------------------------------------------------------------------
__global__ __launch_bounds__(256)
void gdn_scan(const unsigned short* __restrict__ qb,
              const unsigned short* __restrict__ kb,
              const float* __restrict__ vv, const float* __restrict__ gg,
              const float* __restrict__ bb, float* __restrict__ out)
{
    extern __shared__ char smem_raw[];
    float*          ST  = (float*)smem_raw;                   // [64 col][128 dk] f32 state^T
    unsigned short* SbT = (unsigned short*)(ST + 64 * 128);   // [64 col][128 dk] bf16 mirror^T
    unsigned short* Qp  = SbT + 64 * 128;                     // [32 t][128 dk] bf16
    unsigned short* Kp  = Qp + 32 * 128;                      // [32 t][128 dk] bf16
    unsigned short* KdT = Kp + 32 * 128;                      // [128 dk][32 t] bf16 k*e^{G31-Gj}
    unsigned short* DbT = KdT + 128 * 32;                     // [64 n][32 t]  bf16 delta^T
    unsigned short* Ab  = DbT + 64 * 32;                      // [32][32] bf16 scaled q.k Gram
    float* RT = (float*)(Ab + 32 * 32);                       // [64 n][32 t] rhs / delta f32
    float* OT = RT + 64 * 32;                                 // [64 n][32 t] output accum
    float* AM = OT + 64 * 32;                                 // [32][32] k.k Gram (row-major)
    float* AQ = AM + 32 * 32;                                 // [32][32] q.k Gram (row-major)
    float* Gc = AQ + 32 * 32;                                 // 32 cumulative log-decay
    float* Eg = Gc + 32;                                      // exp(G_i)
    float* El = Eg + 32;                                      // exp(G_31 - G_i)
    float* Bt = El + 32;                                      // beta_i

    const int tid  = threadIdx.x;
    const int lane = tid & 31;
    const int wv   = tid >> 5;                // 0..7

    const int blk = blockIdx.x;               // 0..127
    const int b   = blk >> 6;
    const int hv  = (blk >> 1) & 31;
    const int dv0 = (blk & 1) * 64;
    const int hk  = hv >> 1;                  // rep = HV/HK = 2

    for (int i = tid; i < 64 * 128; i += 256) { ST[i] = 0.f; SbT[i] = 0; }
    __syncthreads();

    for (int ch = 0; ch < T_ / 32; ++ch) {
        const int t0 = ch * 32;

        // prefetch next chunk's q/k/v rows one full chunk ahead (L2 -> WGP)
        if (ch + 1 < T_ / 32 && tid < 32) {
            size_t nb = (size_t)(b * T_ + t0 + 32 + tid);
            const unsigned short* pq = qb + nb * 2048 + hk * 128;
            const unsigned short* pk = kb + nb * 2048 + hk * 128;
            const float* pv = vv + nb * 4096 + hv * 128 + dv0;
            __builtin_prefetch(pq, 0, 0);  __builtin_prefetch(pq + 64, 0, 0);
            __builtin_prefetch(pk, 0, 0);  __builtin_prefetch(pk + 64, 0, 0);
            __builtin_prefetch(pv, 0, 0);  __builtin_prefetch(pv + 32, 0, 0);
        }

        if (tid < 32) {
            Gc[tid] = gg[(size_t)(b * T_ + t0 + tid) * HV_ + hv];
            Bt[tid] = bb[(size_t)(b * T_ + t0 + tid) * HV_ + hv];
        }
        __syncthreads();
        if (tid == 0) {
            float acc = 0.f;
#pragma unroll
            for (int i = 0; i < 32; ++i) { acc += Gc[i]; Gc[i] = acc; }
        }
        __syncthreads();
        if (tid < 32) {
            Eg[tid] = __expf(Gc[tid]);
            El[tid] = __expf(Gc[31] - Gc[tid]);
        }
        __syncthreads();

        // stage q/k chunk: 16 contiguous bf16 per thread (32B coalesced global
        // loads, 32B LDS vector stores) + scattered KdT transpose with scale.
        {
            int i0 = tid * 16;                 // 4096 = 256*16
            int r = i0 >> 7, d = i0 & 127;
            size_t base = (size_t)(b * T_ + t0 + r) * 2048 + hk * 128 + d;
            v16u qv = *(const v16u*)(qb + base);
            v16u kv = *(const v16u*)(kb + base);
            *(v16u*)(Qp + i0) = qv;
            *(v16u*)(Kp + i0) = kv;
            float el = El[r];
#pragma unroll
            for (int e = 0; e < 16; ++e)
                KdT[(d + e) * 32 + r] = f2bf(bf2f(kv[e]) * el);
        }
        __syncthreads();

        // --- P = Qp @ S -> OT ;  KS = Kp @ S -> RT  (each wave: one 16x16 tile)
        {
            int mT = wv >> 2, nT = wv & 3;
            v8f accP = {}; v8f accK = {};
#pragma unroll
            for (int kk = 0; kk < 4; ++kk) {
                v16bf bf = load_b_tr(SbT, 128, kk * 32, nT * 16, lane);
                v16bf aQ = load_a(Qp, 128, mT * 16, kk * 32, lane);
                v16bf aK = load_a(Kp, 128, mT * 16, kk * 32, lane);
                accP = wmma_bf16(aQ, bf, accP);
                accK = wmma_bf16(aK, bf, accK);
            }
            store_cT(OT, 32, mT * 16, nT * 16, lane, accP);
            store_cT(RT, 32, mT * 16, nT * 16, lane, accK);
        }
        // --- Grams: waves 0-3: AM = Kp Kp^T ; waves 4-7: AQ = Qp Kp^T
        {
            int j  = wv & 3;
            int mT = j >> 1, nT = j & 1;
            const unsigned short* Abuf = (wv < 4) ? Kp : Qp;
            v8f acc = {};
#pragma unroll
            for (int kk = 0; kk < 4; ++kk) {
                v16bf a  = load_a(Abuf, 128, mT * 16, kk * 32, lane);
                v16bf bt = load_b_tr(Kp, 128, kk * 32, nT * 16, lane); // Kp[n][K]
                acc = wmma_bf16(a, bt, acc);
            }
            store_c_rm((wv < 4) ? AM : AQ, 32, mT * 16, nT * 16, lane, acc);
        }
        __syncthreads();

        // elementwise: R = beta*(v - e^G * KS) ; O = e^G * O   (v coalesced)
        for (int i = tid; i < 2048; i += 256) {
            int t = i >> 6, n = i & 63;
            float v = vv[(size_t)(b * T_ + t0 + t) * 4096 + hv * 128 + dv0 + n];
            int ix = n * 32 + t;
            RT[ix] = Bt[t] * (v - Eg[t] * RT[ix]);
            OT[ix] = Eg[t] * OT[ix];
        }
        // scale Grams: AM_tj = beta_t e^{Gt-Gj}(k.k) strict-lower ; Ab = tril(e^{Gt-Gj} q.k)
        for (int i = tid; i < 1024; i += 256) {
            int r = i >> 5, j = i & 31;
            float e = __expf(Gc[r] - Gc[j]);
            AM[i] = (j < r) ? Bt[r] * e * AM[i] : 0.f;
            Ab[i] = f2bf((j <= r) ? e * AQ[i] : 0.f);
        }
        __syncthreads();

        // forward substitution: delta = (I + AM)^{-1} R ; one column per thread,
        // column n is thread-private contiguous RT[n][0..31].
        if (tid < 64) {
            int n = tid;
            for (int r = 1; r < 32; ++r) {
                float acc = RT[n * 32 + r];
                for (int j = 0; j < r; ++j) acc -= AM[r * 32 + j] * RT[n * 32 + j];
                RT[n * 32 + r] = acc;
            }
            v16u d0, d1;
#pragma unroll
            for (int e = 0; e < 16; ++e) {
                d0[e] = f2bf(RT[n * 32 + e]);
                d1[e] = f2bf(RT[n * 32 + 16 + e]);
            }
            *(v16u*)(DbT + n * 32) = d0;
            *(v16u*)(DbT + n * 32 + 16) = d1;
        }
        __syncthreads();

        // --- intra-chunk output: O += Ab @ Db
        {
            int mT = wv >> 2, nT = wv & 3;
            v16bf a  = load_a(Ab, 32, mT * 16, 0, lane);
            v16bf bf = load_b_tr(DbT, 32, 0, nT * 16, lane);
            v8f c = load_cT(OT, 32, mT * 16, nT * 16, lane);
            c = wmma_bf16(a, bf, c);
            store_cT(OT, 32, mT * 16, nT * 16, lane, c);
        }
        __syncthreads();

        // write output (coalesced), rescale state by e^{G_31}
        const float eAll = __expf(Gc[31]);
        for (int i = tid; i < 2048; i += 256) {
            int t = i >> 6, n = i & 63;
            out[(size_t)(b * T_ + t0 + t) * 4096 + hv * 128 + dv0 + n] = OT[n * 32 + t];
        }
        for (int i = tid; i < 64 * 128; i += 256) ST[i] *= eAll;
        __syncthreads();

        // --- state update: S += Kd^T @ Db ; refresh bf16 mirror (1x b128)
#pragma unroll
        for (int job = wv; job < 32; job += 8) {
            int mT = job >> 2, nT = job & 3;
            v16bf a  = load_a(KdT, 32, mT * 16, 0, lane);      // A[m][K]=k_K[dk]*El[K]
            v16bf bf = load_b_tr(DbT, 32, 0, nT * 16, lane);
            v8f c = load_cT(ST, 128, mT * 16, nT * 16, lane);
            c = wmma_bf16(a, bf, c);
            store_cT(ST, 128, mT * 16, nT * 16, lane, c);
            int col = nT * 16 + (lane & 15);
            int rbase = mT * 16 + ((lane >> 4) ? 8 : 0);
            v8u sb;
#pragma unroll
            for (int r2 = 0; r2 < 8; ++r2) sb[r2] = f2bf(c[r2]);
            *(v8u*)(SbT + col * 128 + rbase) = sb;
        }
        __syncthreads();
    }
}

// ---------------------------------------------------------------------------
extern "C" void kernel_launch(void* const* d_in, const int* in_sizes, int n_in,
                              void* d_out, int out_size, void* d_ws, size_t ws_size,
                              hipStream_t stream) {
    (void)in_sizes; (void)n_in; (void)out_size; (void)ws_size;
    const float* x    = (const float*)d_in[0];   // mixed_qkv
    const float* bIn  = (const float*)d_in[1];   // b
    const float* aIn  = (const float*)d_in[2];   // a
    const float* cw   = (const float*)d_in[3];   // conv_w
    const float* dtb  = (const float*)d_in[4];   // dt_bias
    const float* alog = (const float*)d_in[5];   // A_log
    float* out = (float*)d_out;

    // workspace (~102 MB): v f32, gates f32, q/k bf16
    float* vv = (float*)d_ws;
    float* gg = vv + (size_t)B_ * T_ * HV_ * DV_;
    float* bb = gg + (size_t)B_ * T_ * HV_;
    unsigned short* qbuf = (unsigned short*)(bb + (size_t)B_ * T_ * HV_);
    unsigned short* kbuf = qbuf + (size_t)B_ * T_ * HK_ * DK_;

    gdn_preprocess<<<B_ * T_, 256, 0, stream>>>(x, bIn, aIn, cw, dtb, alog,
                                                qbuf, kbuf, vv, gg, bb);

    const size_t smem =
        64 * 128 * 4 +            // ST
        64 * 128 * 2 +            // SbT
        32 * 128 * 2 * 2 +        // Qp, Kp
        128 * 32 * 2 +            // KdT
        64 * 32 * 2 +             // DbT
        32 * 32 * 2 +             // Ab
        64 * 32 * 4 * 2 +         // RT, OT
        32 * 32 * 4 * 2 +         // AM, AQ
        32 * 4 * 4;               // Gc, Eg, El, Bt   => 104960 bytes
    gdn_scan<<<B_ * HV_ * 2, 256, smem, stream>>>(qbuf, kbuf, vv, gg, bb, out);
}
